// KMeans_SingleImage_10763188044353
// MI455X (gfx1250) — matrix operand order repaired and checked by reading, not compiled, not run on hardware
//
#include <hip/hip_runtime.h>
#include <hip/hip_bf16.h>

// ---------------------------------------------------------------------------
// Cosine k-means (K=64, 30 iters) on X=[9216,1024] + bilinear 14x upsample.
// GEMMs use v_wmma_f32_16x16x32_bf16 (CDNA5 wave32 WMMA).
// ---------------------------------------------------------------------------

#define N_TOK   9216        // 96*96 tokens
#define C_DIM   1024        // channels
#define K_CL    64          // clusters
#define HP      96
#define OUT_HW  1344        // 96*14
#define N_ITERS 30
#define TOL     1e-4f
#define EPSN    1e-10f

typedef __bf16 v16bf __attribute__((ext_vector_type(16)));
typedef float  v8f   __attribute__((ext_vector_type(8)));

union ABFrag { v16bf v; unsigned int u[8]; };

__device__ __forceinline__ unsigned short f32_to_bf16(float f) {
    unsigned int u = __float_as_uint(f);
    unsigned int r = u + 0x7FFFu + ((u >> 16) & 1u);   // round-to-nearest-even
    return (unsigned short)(r >> 16);
}

__device__ __forceinline__ float block_reduce_sum_256(float v, float* sm) {
    int t = threadIdx.x;
    sm[t] = v; __syncthreads();
    for (int s = 128; s > 0; s >>= 1) { if (t < s) sm[t] += sm[t + s]; __syncthreads(); }
    float r = sm[0]; __syncthreads();
    return r;
}

// --- init: centers0 = strided rows of X, zero shift slots + done flag --------
__global__ void init_kernel(const float* __restrict__ feat, float* __restrict__ centers,
                            float* __restrict__ shift, int* __restrict__ done) {
    int k = blockIdx.x;                 // cluster
    int srcRow = k * (N_TOK / K_CL);    // deterministic "random" init row
    for (int i = 0; i < 4; ++i) {
        int c = threadIdx.x + i * 256;
        centers[k * C_DIM + c] = feat[(size_t)c * N_TOK + srcRow];
    }
    if (blockIdx.x == 0) {
        if (threadIdx.x < N_ITERS) shift[threadIdx.x] = 0.0f;
        if (threadIdx.x == 32) *done = 0;
    }
}

// --- per-token L2 norms (coalesced over tokens) ------------------------------
__global__ void row_norms_kernel(const float* __restrict__ feat, float* __restrict__ norms) {
    int n = blockIdx.x * 256 + threadIdx.x;   // token id
    float s = 0.0f;
    #pragma unroll 4
    for (int c = 0; c < C_DIM; ++c) {
        float v = feat[(size_t)c * N_TOK + n];
        s += v * v;
    }
    norms[n] = sqrtf(s);
}

// --- transpose [C][N] -> Xt fp32 [N][C] and Xn bf16 [N][C] (normalized) ------
__global__ void transpose_kernel(const float* __restrict__ feat, const float* __restrict__ norms,
                                 float* __restrict__ Xt, unsigned short* __restrict__ Xn) {
    __shared__ float tile[32][33];
    int n0 = blockIdx.x * 32;
    int c0 = blockIdx.y * 32;
    int tx = threadIdx.x, ty = threadIdx.y;   // 32 x 8
    #pragma unroll
    for (int r = 0; r < 4; ++r) {
        int c = c0 + ty + r * 8;
        int n = n0 + tx;
        tile[ty + r * 8][tx] = feat[(size_t)c * N_TOK + n];   // coalesced on n
    }
    __syncthreads();
    #pragma unroll
    for (int r = 0; r < 4; ++r) {
        int n = n0 + ty + r * 8;
        int c = c0 + tx;
        float v = tile[tx][ty + r * 8];
        float inv = 1.0f / fmaxf(norms[n], EPSN);
        Xt[(size_t)n * C_DIM + c] = v;                        // coalesced on c
        Xn[(size_t)n * C_DIM + c] = f32_to_bf16(v * inv);
    }
}

// --- normalize centers -> Cn bf16 [64][1024] ---------------------------------
__global__ void center_norm_kernel(const float* __restrict__ centers,
                                   unsigned short* __restrict__ Cn) {
    __shared__ float sm[256];
    int k = blockIdx.x;
    float s = 0.0f;
    #pragma unroll
    for (int i = 0; i < 4; ++i) {
        float v = centers[k * C_DIM + threadIdx.x + i * 256];
        s += v * v;
    }
    float nrm = sqrtf(block_reduce_sum_256(s, sm));
    float inv = 1.0f / fmaxf(nrm, EPSN);
    #pragma unroll
    for (int i = 0; i < 4; ++i) {
        int c = threadIdx.x + i * 256;
        Cn[k * C_DIM + c] = f32_to_bf16(centers[k * C_DIM + c] * inv);
    }
}

// --- shared WMMA tile: S[16x16] at (m0, n0), K=1024 via 32 wmma steps --------
__device__ __forceinline__ v8f wmma_tile(const unsigned short* __restrict__ Xn,
                                         const unsigned short* __restrict__ Cn,
                                         int m0, int n0, int lm, int hi) {
    v8f acc = {0.0f, 0.0f, 0.0f, 0.0f, 0.0f, 0.0f, 0.0f, 0.0f};
    const unsigned int* arow = (const unsigned int*)(Xn + (size_t)(m0 + lm) * C_DIM);
    const unsigned int* brow = (const unsigned int*)(Cn + (size_t)(n0 + lm) * C_DIM);
    for (int ks = 0; ks < C_DIM / 32; ++ks) {
        int kb = ks * 32;
        ABFrag a, b;
        #pragma unroll
        for (int v = 0; v < 8; ++v) {
            // A 16x32 bf16 layout: VGPR v -> K = (v&4?16:0) + (v&3)*2 (+8 for hi lanes)
            int ka = kb + ((v & 4) ? 16 : 0) + (v & 3) * 2 + (hi ? 8 : 0);
            a.u[v] = arow[ka >> 1];
            // B 32x16 bf16 layout: VGPR v -> K = (v&4?16:0) + (v&3)*2 (+8 for hi lanes)
            int kB = kb + ((v & 4) ? 16 : 0) + (v & 3) * 2 + (hi ? 8 : 0);
            b.u[v] = brow[kB >> 1];
        }
        acc = __builtin_amdgcn_wmma_f32_16x16x32_bf16(
            false, a.v, false, b.v, (short)0, acc, false, false);
    }
    return acc;
}

// --- assignment: argmax_n cos(x, c_n); 4 waves cover all 64 clusters ---------
__global__ __launch_bounds__(128) void assign_kernel(const unsigned short* __restrict__ Xn,
                                                     const unsigned short* __restrict__ Cn,
                                                     int* __restrict__ assign) {
    __shared__ float S[16][64];
    int tid  = threadIdx.x;
    int wave = tid >> 5, lane = tid & 31;
    int lm = lane & 15, hi = lane >> 4;
    int m0 = blockIdx.x * 16;
    int n0 = wave * 16;
    v8f acc = wmma_tile(Xn, Cn, m0, n0, lm, hi);
    #pragma unroll
    for (int r = 0; r < 8; ++r) S[r + hi * 8][n0 + lm] = acc[r];
    __syncthreads();
    if (tid < 16) {
        float best = -3.4e38f; int bk = 0;
        for (int k = 0; k < K_CL; ++k) {
            float v = S[tid][k];
            if (v > best) { best = v; bk = k; }     // first-max, matches argmin/argmax
        }
        assign[m0 + tid] = bk;
    }
}

// --- segment sums: block owns 64 channels exclusively (LDS partials) ---------
__global__ __launch_bounds__(256) void segsum_kernel(const float* __restrict__ Xt,
                                                     const int* __restrict__ assign,
                                                     float* __restrict__ sums,
                                                     float* __restrict__ counts) {
    __shared__ float lsum[K_CL * 64];
    __shared__ int   lcnt[K_CL];
    int tid = threadIdx.x;
    int cbase = blockIdx.x * 64;
    int coff = tid & 63;
    int rs   = tid >> 6;                 // 4 parallel row streams
    #pragma unroll
    for (int i = 0; i < 16; ++i) lsum[i * 256 + tid] = 0.0f;
    if (tid < K_CL) lcnt[tid] = 0;
    __syncthreads();
    for (int n = rs; n < N_TOK; n += 4) {
        int a = assign[n];
        atomicAdd(&lsum[a * 64 + coff], Xt[(size_t)n * C_DIM + cbase + coff]);
        if (blockIdx.x == 0 && coff == 0) atomicAdd(&lcnt[a], 1);
    }
    __syncthreads();
    #pragma unroll
    for (int i = 0; i < 16; ++i) {
        int e = i * 256 + tid;
        int kk = e >> 6, jj = e & 63;
        sums[kk * C_DIM + cbase + jj] = lsum[e];
    }
    if (blockIdx.x == 0 && tid < K_CL) counts[tid] = (float)lcnt[tid];
}

// --- new centers + per-cluster shift contribution ----------------------------
__global__ void newcenters_kernel(const float* __restrict__ sums, const float* __restrict__ counts,
                                  const float* __restrict__ centers, float* __restrict__ newC,
                                  float* __restrict__ shiftSlot) {
    __shared__ float sm[256];
    int k = blockIdx.x;
    float cnt = counts[k];
    float sq = 0.0f;
    #pragma unroll
    for (int i = 0; i < 4; ++i) {
        int c = threadIdx.x + i * 256;
        float oldv = centers[k * C_DIM + c];
        float v = (cnt > 0.0f) ? (sums[k * C_DIM + c] / fmaxf(cnt, 1.0f)) : oldv;
        newC[k * C_DIM + c] = v;
        float d = v - oldv;
        sq += d * d;
    }
    float tot = block_reduce_sum_256(sq, sm);
    if (threadIdx.x == 0) atomicAdd(shiftSlot, sqrtf(tot));
}

// --- freeze-after-convergence: centers = done ? centers : newC ---------------
__global__ void freeze_kernel(float* __restrict__ centers, const float* __restrict__ newC,
                              const float* __restrict__ shiftSlot, int* __restrict__ done) {
    int dn = *done;
    __syncthreads();
    if (!dn) {
        for (int i = threadIdx.x; i < K_CL * C_DIM; i += 256) centers[i] = newC[i];
    }
    __syncthreads();
    if (threadIdx.x == 0) {
        float s = *shiftSlot;
        if (s * s < TOL) *done = 1;
    }
}

// --- final low-res cosine logits [9216][64] ---------------------------------
__global__ __launch_bounds__(128) void logits_kernel(const unsigned short* __restrict__ Xn,
                                                     const unsigned short* __restrict__ Cn,
                                                     float* __restrict__ lowres) {
    int tid  = threadIdx.x;
    int wave = tid >> 5, lane = tid & 31;
    int lm = lane & 15, hi = lane >> 4;
    int m0 = blockIdx.x * 16;
    int n0 = wave * 16;
    v8f acc = wmma_tile(Xn, Cn, m0, n0, lm, hi);
    #pragma unroll
    for (int r = 0; r < 8; ++r) {
        int m = m0 + r + hi * 8;
        lowres[(size_t)m * K_CL + n0 + lm] = acc[r];
    }
}

// --- fused bilinear 14x upsample + channel argmax ----------------------------
__global__ __launch_bounds__(256) void upsample_kernel(const float* __restrict__ lowres,
                                                       float* __restrict__ out) {
    const int NP = OUT_HW * OUT_HW;
    int p = blockIdx.x * 256 + threadIdx.x;
    int y = p / OUT_HW, x = p - y * OUT_HW;
    const float scl = (float)HP / (float)OUT_HW;   // 1/14, half-pixel mapping
    float sy = (y + 0.5f) * scl - 0.5f;
    float sx = (x + 0.5f) * scl - 0.5f;
    float fy = floorf(sy), fx = floorf(sx);
    float wy = sy - fy,    wx = sx - fx;
    int y0 = (int)fy, x0 = (int)fx;
    int y1 = y0 + 1,  x1 = x0 + 1;
    y0 = min(max(y0, 0), HP - 1); y1 = min(max(y1, 0), HP - 1);
    x0 = min(max(x0, 0), HP - 1); x1 = min(max(x1, 0), HP - 1);
    float w00 = (1.0f - wy) * (1.0f - wx);
    float w01 = (1.0f - wy) * wx;
    float w10 = wy * (1.0f - wx);
    float w11 = wy * wx;
    const float* p00 = lowres + (size_t)(y0 * HP + x0) * K_CL;
    const float* p01 = lowres + (size_t)(y0 * HP + x1) * K_CL;
    const float* p10 = lowres + (size_t)(y1 * HP + x0) * K_CL;
    const float* p11 = lowres + (size_t)(y1 * HP + x1) * K_CL;
    float best = -3.4e38f; int bk = 0;
    #pragma unroll 4
    for (int k = 0; k < K_CL; ++k) {
        float v = w00 * p00[k] + w01 * p01[k] + w10 * p10[k] + w11 * p11[k];
        out[(size_t)NP + (size_t)k * NP + p] = v;   // logits_up, coalesced per k
        if (v > best) { best = v; bk = k; }
    }
    out[p] = (float)bk;                              // cluster_ids (as float)
}

// ---------------------------------------------------------------------------
extern "C" void kernel_launch(void* const* d_in, const int* in_sizes, int n_in,
                              void* d_out, int out_size, void* d_ws, size_t ws_size,
                              hipStream_t stream) {
    (void)in_sizes; (void)n_in; (void)ws_size;
    const float* feat = (const float*)d_in[0];
    float* out = (float*)d_out;
    float* ws  = (float*)d_ws;

    // ws layout (floats); total ~3.4 MB
    float* centers = ws;                               // 64*1024
    float* newC    = centers + K_CL * C_DIM;           // 64*1024
    float* sums    = newC    + K_CL * C_DIM;           // 64*1024
    float* norms   = sums    + K_CL * C_DIM;           // 9216
    int*   assign  = (int*)(norms + N_TOK);            // 9216
    float* counts  = (float*)(assign + N_TOK);         // 64
    float* shift   = counts + K_CL;                    // 32 slots
    int*   done    = (int*)(shift + 32);               // 1
    unsigned short* Cn = (unsigned short*)(done + 8);  // 64*1024 bf16
    float* lowres  = (float*)(Cn + K_CL * C_DIM);      // 9216*64

    // big scratch lives in the tail of d_out (overwritten by final upsample)
    size_t XT_OFF = (size_t)out_size - (size_t)N_TOK * C_DIM;           // fp32 [N][C]
    size_t XN_OFF = XT_OFF - (size_t)N_TOK * C_DIM / 2;                 // bf16 [N][C]
    float*          Xt = out + XT_OFF;
    unsigned short* Xn = (unsigned short*)(out + XN_OFF);

    init_kernel<<<K_CL, 256, 0, stream>>>(feat, centers, shift, done);
    row_norms_kernel<<<N_TOK / 256, 256, 0, stream>>>(feat, norms);
    transpose_kernel<<<dim3(N_TOK / 32, C_DIM / 32), dim3(32, 8), 0, stream>>>(feat, norms, Xt, Xn);

    for (int it = 0; it < N_ITERS; ++it) {
        center_norm_kernel<<<K_CL, 256, 0, stream>>>(centers, Cn);
        assign_kernel<<<N_TOK / 16, 128, 0, stream>>>(Xn, Cn, assign);
        segsum_kernel<<<C_DIM / 64, 256, 0, stream>>>(Xt, assign, sums, counts);
        newcenters_kernel<<<K_CL, 256, 0, stream>>>(sums, counts, centers, newC, shift + it);
        freeze_kernel<<<1, 256, 0, stream>>>(centers, newC, shift + it, done);
    }

    center_norm_kernel<<<K_CL, 256, 0, stream>>>(centers, Cn);
    logits_kernel<<<N_TOK / 16, 128, 0, stream>>>(Xn, Cn, lowres);
    upsample_kernel<<<(OUT_HW * OUT_HW) / 256, 256, 0, stream>>>(lowres, out);
}
